// MultiHeadAttention_11364483465730
// MI455X (gfx1250) — matrix operand order repaired and checked by reading, not compiled, not run on hardware
//
#include <hip/hip_runtime.h>
#include <hip/hip_bf16.h>

typedef __attribute__((ext_vector_type(16))) _Float16 v16h;
typedef __attribute__((ext_vector_type(8)))  _Float16 v8h;
typedef __attribute__((ext_vector_type(4)))  _Float16 v4h;
typedef __attribute__((ext_vector_type(8)))  float    v8f;
typedef __attribute__((ext_vector_type(4)))  float    v4f;

// exact pointee type the async-LDS builtin expects (gcc-vector int4)
typedef int i32x4 __attribute__((vector_size(16)));
typedef __attribute__((address_space(1))) i32x4 gas_i32x4;   // global (AS1)
typedef __attribute__((address_space(3))) i32x4 las_i32x4;   // LDS (AS3)

#define NTOK      4096        // N * L
#define DM        1024        // embed dim / d_model
#define ATTN_MASS 2048.0f     // sum over (q,k) of softmax == Lq exactly

// ---------------------------------------------------------------------------
// gfx1250 async global->LDS copy (ASYNCcnt path), with inline-asm fallback.
// LDS offset = low 32 bits of the flat pointer (aperture mapping, ISA 10.2).
// ---------------------------------------------------------------------------
__device__ __forceinline__ void async_copy_b128(const void* gsrc, void* ldst) {
#if __has_builtin(__builtin_amdgcn_global_load_async_to_lds_b128)
    __builtin_amdgcn_global_load_async_to_lds_b128(
        (gas_i32x4*)(uintptr_t)gsrc,
        (las_i32x4*)(unsigned)(uintptr_t)ldst,
        0, 0);
#else
    unsigned loff = (unsigned)(uintptr_t)ldst;
    asm volatile("global_load_async_to_lds_b128 %0, %1, off"
                 :: "v"(loff), "v"(gsrc) : "memory");
#endif
}

__device__ __forceinline__ void wait_async0() {
#if __has_builtin(__builtin_amdgcn_s_wait_asynccnt)
    __builtin_amdgcn_s_wait_asynccnt(0);
#else
    asm volatile("s_wait_asynccnt 0x0" ::: "memory");
#endif
}

// ---------------------------------------------------------------------------
// Kernel 1: fold 2048 * BlockDiag(Wv^T) @ Wo^T into one 1024x1024 f16 matrix.
// M[e][d] = 2048 * sum_t Wo[e, (d/64)*64 + t] * Wv[t, d%64]
// ---------------------------------------------------------------------------
__global__ __launch_bounds__(256)
void mha_fuse_weights_kernel(const float* __restrict__ Wv,
                             const float* __restrict__ Wo,
                             _Float16* __restrict__ M) {
    int idx = blockIdx.x * 256 + threadIdx.x;   // 0 .. 1M-1
    int e = idx >> 10;
    int d = idx & 1023;
    const float* wo = Wo + e * 1024 + ((d >> 6) << 6);
    const float* wv = Wv + (d & 63);
    float acc = 0.0f;
#pragma unroll 8
    for (int t = 0; t < 64; ++t)
        acc += wo[t] * wv[t * 64];
    M[idx] = (_Float16)(acc * ATTN_MASS);
}

// ---------------------------------------------------------------------------
// Kernel 2: b_comb[e] = 2048 * sum_c bv[c%64] * Wo[e,c] + bo[e]
// ---------------------------------------------------------------------------
__global__ __launch_bounds__(256)
void mha_fuse_bias_kernel(const float* __restrict__ Wo,
                          const float* __restrict__ bv,
                          const float* __restrict__ bo,
                          float* __restrict__ bias) {
    int e = blockIdx.x * 256 + threadIdx.x;     // 0..1023
    const float* wo = Wo + e * 1024;
    float acc = 0.0f;
    for (int c = 0; c < 1024; ++c)
        acc += bv[c & 63] * wo[c];
    bias[e] = acc * ATTN_MASS + bo[e];
}

// ---------------------------------------------------------------------------
// Kernel 3: out[r,e] = sum_d V[r,d] * M[e,d] + bias[e]
// 64x64 block tile, 4 wave32s; each wave: 16 rows x 64 cols = 4 WMMA tiles.
// K stepped by 32; double-buffered LDS; B tile via async global->LDS DMA,
// A tile via register path with fp32->f16 convert in flight.
// ---------------------------------------------------------------------------
#define LDS_STRIDE 40   // 32 halves data + pad (80B rows -> conflict-free gathers)
#define KSTEPS     32   // 1024 / 32

__device__ __forceinline__ void issue_async_B(const _Float16* M, int colBase,
                                              int k0, int tid, _Float16* sB) {
#pragma unroll
    for (int i = 0; i < 2; ++i) {
        int c = tid + i * 128;          // 256 x 16B chunks
        int r = c >> 2, s = c & 3;
        async_copy_b128(M + (colBase + r) * 1024 + k0 + s * 8,
                        sB + r * LDS_STRIDE + s * 8);
    }
}

__device__ __forceinline__ void load_A(const float* V, int rowBase, int k0,
                                       int tid, v4f* areg) {
#pragma unroll
    for (int i = 0; i < 4; ++i) {
        int c = tid + i * 128;          // 512 x float4 chunks
        int r = c >> 3, s = c & 7;
        areg[i] = *(const v4f*)(V + (rowBase + r) * 1024 + k0 + s * 4);
    }
}

__device__ __forceinline__ void store_A(const v4f* areg, int tid, _Float16* sA) {
#pragma unroll
    for (int i = 0; i < 4; ++i) {
        int c = tid + i * 128;
        int r = c >> 3, s = c & 7;
        v4h ah;
        ah[0] = (_Float16)areg[i][0];
        ah[1] = (_Float16)areg[i][1];
        ah[2] = (_Float16)areg[i][2];
        ah[3] = (_Float16)areg[i][3];
        *(v4h*)(sA + r * LDS_STRIDE + s * 4) = ah;
    }
}

__global__ __launch_bounds__(128)
void mha_fused_gemm_kernel(const float* __restrict__ V,
                           const _Float16* __restrict__ M,
                           const float* __restrict__ bias,
                           float* __restrict__ out) {
    __shared__ _Float16 sA[2][64 * LDS_STRIDE];
    __shared__ _Float16 sB[2][64 * LDS_STRIDE];

    const int tid  = threadIdx.x;
    const int lane = tid & 31;
    const int wave = tid >> 5;
    const int l    = lane & 15;
    const int hiK  = lane >> 4;

    const int rowBase = blockIdx.y * 64;   // token rows
    const int colBase = blockIdx.x * 64;   // output features

    v8f acc[4] = {};
    v4f areg[4];

    // prologue: stage tile 0 into buffer 0
    issue_async_B(M, colBase, 0, tid, sB[0]);
    load_A(V, rowBase, 0, tid, areg);
    store_A(areg, tid, sA[0]);
    wait_async0();
    __syncthreads();

    for (int it = 0; it < KSTEPS; ++it) {
        const int  buf  = it & 1;
        const bool more = (it + 1) < KSTEPS;

        if (more) {
            // kick next tile's DMA + global loads before computing this one
            issue_async_B(M, colBase, (it + 1) * 32, tid, sB[buf ^ 1]);
            load_A(V, rowBase, (it + 1) * 32, tid, areg);
            if (it + 2 < KSTEPS) {   // -> global_prefetch_b8
                __builtin_prefetch(V + (rowBase + (tid >> 3)) * 1024 +
                                   (it + 2) * 32 + (tid & 7) * 4, 0, 0);
                __builtin_prefetch(M + (colBase + (tid >> 2)) * 1024 +
                                   (it + 2) * 32 + (tid & 3) * 8, 0, 0);
            }
        }

        // ---- A fragment (16x32 f16): lanes 0-15 K{0..7,16..23},
        //      lanes 16-31 K{8..15,24..31}
        const _Float16* aRow = &sA[buf][(wave * 16 + l) * LDS_STRIDE];
        v8h alo = *(const v8h*)(aRow + hiK * 8);
        v8h ahi = *(const v8h*)(aRow + 16 + hiK * 8);
        v16h a = __builtin_shufflevector(alo, ahi,
                 0,1,2,3,4,5,6,7,8,9,10,11,12,13,14,15);

        // ---- B fragments first, then 4 back-to-back WMMAs
        v16h b[4];
#pragma unroll
        for (int ct = 0; ct < 4; ++ct) {
            const _Float16* bRow = &sB[buf][(ct * 16 + l) * LDS_STRIDE + hiK * 8];
            v8h blo = *(const v8h*)(bRow);
            v8h bhi = *(const v8h*)(bRow + 16);
            b[ct] = __builtin_shufflevector(blo, bhi,
                    0,1,2,3,4,5,6,7,8,9,10,11,12,13,14,15);
        }
#pragma unroll
        for (int ct = 0; ct < 4; ++ct)
            acc[ct] = __builtin_amdgcn_wmma_f32_16x16x32_f16(
                false, a, false, b[ct], (short)0, acc[ct], false, false);

        if (more) {
            store_A(areg, tid, sA[buf ^ 1]);   // cvt + ds_store after WMMAs issued
            wait_async0();                      // drain our ASYNCcnt (B DMA)
            __syncthreads();                    // one barrier per K-step
        }
    }

    // ---- writeback: f32 C/D layout (VGPR v -> row hiK*8+v, col l) ---------
#pragma unroll
    for (int ct = 0; ct < 4; ++ct) {
        int col = colBase + ct * 16 + l;
        float bc = bias[col];
#pragma unroll
        for (int v = 0; v < 8; ++v) {
            int row = rowBase + wave * 16 + hiK * 8 + v;
            out[row * 1024 + col] = acc[ct][v] + bc;
        }
    }
}

// ---------------------------------------------------------------------------
// launcher
// inputs: 0:Q 1:K 2:V 3:Wq 4:bq 5:Wk 6:bk 7:Wv 8:bv 9:Wo 10:bo (all fp32)
// ---------------------------------------------------------------------------
extern "C" void kernel_launch(void* const* d_in, const int* in_sizes, int n_in,
                              void* d_out, int out_size, void* d_ws, size_t ws_size,
                              hipStream_t stream) {
    const float* V  = (const float*)d_in[2];
    const float* Wv = (const float*)d_in[7];
    const float* bv = (const float*)d_in[8];
    const float* Wo = (const float*)d_in[9];
    const float* bo = (const float*)d_in[10];
    float* out = (float*)d_out;

    _Float16* M   = (_Float16*)d_ws;                        // 2 MB
    float*    bcb = (float*)((char*)d_ws + (1u << 21));     // 1024 * 4B

    mha_fuse_weights_kernel<<<dim3((DM * DM) / 256), 256, 0, stream>>>(Wv, Wo, M);
    mha_fuse_bias_kernel<<<dim3(DM / 256), 256, 0, stream>>>(Wo, bv, bo, bcb);
    mha_fused_gemm_kernel<<<dim3(DM / 64, NTOK / 64), 128, 0, stream>>>(V, M, bcb, out);
}
